// self_LSTM_truncated_90907277787362
// MI455X (gfx1250) — compile-verified, matrix operand order checked by reading, not run on hardware
//
#include <hip/hip_runtime.h>
#include <hip/hip_bf16.h>

// Problem constants (match reference)
#define B_   128
#define T_   256
#define H_   128
#define H4_  512
#define C_   10
#define TP1  257

typedef __attribute__((ext_vector_type(16))) __bf16 v16bf;
typedef __attribute__((ext_vector_type(8)))  __bf16 v8bf;
typedef __attribute__((ext_vector_type(8)))  float  v8f;

union AB16 { v16bf v; v8bf h[2]; };

__device__ __forceinline__ float sigmf(float v) { return 1.0f / (1.0f + __expf(-v)); }

// Dynamic-LDS partition offsets (bytes). Total 175168 B <= 320 KB WGP LDS.
#define LDS_WBF   0                       // [4H][H] bf16            131072 B
#define LDS_HBF   131072                  // [B][H]  bf16             32768 B
#define LDS_XCOL  163840                  // [B]     f32                512 B
#define LDS_S1    164352                  // [B]     f32                512 B
#define LDS_WIH   164864                  // [4H]    f32               2048 B
#define LDS_BSUM  166912                  // [4H]    f32               2048 B
#define LDS_WT1   168960                  // [H]     f32                512 B
#define LDS_WT2   169472                  // [H]     f32                512 B
#define LDS_FCW   169984                  // [C][H]  f32               5120 B
#define LDS_FCB   175104                  // [C]     f32                 64 B
#define LDS_TOTAL 175168

// ---------------------------------------------------------------------------
// Persistent kernel: one workgroup (32 waves) iterates all T steps.
//   Phase A: gates = h @ W_hh^T via v_wmma_f32_16x16x32_bf16 (A and B both
//            served from LDS; per-tile base addresses precomputed so every
//            fragment load is ds_load_b128 base+smallimm, double-buffered
//            so the ds loads overlap the WMMAs), then the LSTM cell update.
//   Phase B: s1[b], s2[t+1][b] attention partial scores (shuffle reduce).
//   Phase C: attention weighted history sum (L2-resident) + FC head.
// ---------------------------------------------------------------------------
__global__ __launch_bounds__(1024)
void lstm_attn_persistent(const float* __restrict__ x,      // [B,T]
                          const float* __restrict__ W_ih,   // [4H]
                          const float* __restrict__ W_hh,   // [4H,H]
                          const float* __restrict__ b_ih,   // [4H]
                          const float* __restrict__ b_hh,   // [4H]
                          const float* __restrict__ w_t,    // [2H]
                          const float* __restrict__ fc_W,   // [C,H]
                          const float* __restrict__ fc_b,   // [C]
                          float* __restrict__ s2g,          // [TP1,B]
                          float* __restrict__ buf,          // [TP1,B,H]
                          float* __restrict__ out)          // [B,T,C]
{
  extern __shared__ __align__(16) char smem[];
  __bf16* wlds   = (__bf16*)(smem + LDS_WBF);
  __bf16* h_bf   = (__bf16*)(smem + LDS_HBF);
  float*  xcol   = (float*) (smem + LDS_XCOL);
  float*  s1s    = (float*) (smem + LDS_S1);
  float*  wih_s  = (float*) (smem + LDS_WIH);
  float*  bsum_s = (float*) (smem + LDS_BSUM);
  float*  wt1_s  = (float*) (smem + LDS_WT1);
  float*  wt2_s  = (float*) (smem + LDS_WT2);
  float*  fcW_s  = (float*) (smem + LDS_FCW);
  float*  fcb_s  = (float*) (smem + LDS_FCB);

  const int tid  = threadIdx.x;
  const int wave = tid >> 5;
  const int lane = tid & 31;
  const int mtile = wave & 7;        // 8 M-tiles of 16 batch rows
  const int jq    = wave >> 3;       // 4 column quarters (32 hh cols each)
  const int l15   = lane & 15;
  const int lhalf = lane >> 4;
  const int mrow  = mtile * 16 + l15;   // A-matrix row = batch b

  // ---- one-time init: stage weights into LDS ----
  for (int i = tid; i < H4_ * H_; i += 1024) wlds[i] = (__bf16)W_hh[i];
  for (int i = tid; i < B_ * H_; i += 1024) h_bf[i] = (__bf16)0.0f;
  for (int i = tid; i < H4_; i += 1024) {
    wih_s[i]  = W_ih[i];
    bsum_s[i] = b_ih[i] + b_hh[i];
  }
  if (tid < H_) { wt1_s[tid] = w_t[tid]; wt2_s[tid] = w_t[H_ + tid]; }
  for (int i = tid; i < C_ * H_; i += 1024) fcW_s[i] = fc_W[i];
  if (tid < C_) fcb_s[tid] = fc_b[tid];

  // Per-tile LDS base addresses (live in VGPRs across the whole t-loop).
  // B-tile i -> N-tile nt = (i>>1)*8 + 2*jq + (i&1); per-lane row (nt*16+l15),
  // per-lane K base (lhalf*8). Fragment loads are base + {kk*64, kk*64+32} B.
  const __bf16* abase = h_bf + mrow * H_ + (lhalf << 3);
  const __bf16* bbase[8];
  #pragma unroll
  for (int i = 0; i < 8; ++i) {
    const int nt = (i >> 1) * 8 + 2 * jq + (i & 1);
    bbase[i] = wlds + (nt * 16 + l15) * H_ + (lhalf << 3);
  }

  float creg[16];                    // cell state, 2 col-tiles x 8 rows
  #pragma unroll
  for (int i = 0; i < 16; ++i) creg[i] = 0.0f;

  const v8f vzero = {0.f, 0.f, 0.f, 0.f, 0.f, 0.f, 0.f, 0.f};

  // fragment load helpers (kk/i become compile-time after full unroll)
  auto loadA = [&](AB16& f, int kk) {
    const __bf16* p = abase + kk * 32;
    f.h[0] = *(const v8bf*)p;
    f.h[1] = *(const v8bf*)(p + 16);
  };
  auto loadB = [&](AB16& f, int kk, int i) {
    const __bf16* p = bbase[i] + kk * 32;
    f.h[0] = *(const v8bf*)p;
    f.h[1] = *(const v8bf*)(p + 16);
  };

  __syncthreads();

  for (int t = 0; t < T_; ++t) {
    // ------------------ Phase A: WMMA GEMM + LSTM cell ------------------
    if (tid < B_) xcol[tid] = x[(size_t)tid * T_ + t];

    v8f acc[4][2];
    #pragma unroll
    for (int q = 0; q < 4; ++q) { acc[q][0] = vzero; acc[q][1] = vzero; }

    AB16 aF[2], bF[2];
    loadA(aF[0], 0);
    loadB(bF[0], 0, 0);
    #pragma unroll
    for (int kk = 0; kk < 4; ++kk) {
      #pragma unroll
      for (int i = 0; i < 8; ++i) {
        // software pipeline: issue next fragment's loads before this WMMA
        if (i < 7) {
          loadB(bF[(i + 1) & 1], kk, i + 1);
        } else if (kk < 3) {
          loadA(aF[(kk + 1) & 1], kk + 1);
          loadB(bF[(i + 1) & 1], kk + 1, 0);
        }
        acc[i >> 1][i & 1] = __builtin_amdgcn_wmma_f32_16x16x32_bf16(
            false, aF[kk & 1].v, false, bF[i & 1].v, (short)0,
            acc[i >> 1][i & 1], false, false);
      }
    }
    __syncthreads();   // all reads of old h_bf done before overwriting

    #pragma unroll
    for (int p = 0; p < 2; ++p) {
      const int hh = (2 * jq + p) * 16 + l15;      // hidden column
      const float wi0 = wih_s[hh],        bs0 = bsum_s[hh];
      const float wi1 = wih_s[H_ + hh],   bs1 = bsum_s[H_ + hh];
      const float wi2 = wih_s[2*H_ + hh], bs2 = bsum_s[2*H_ + hh];
      const float wi3 = wih_s[3*H_ + hh], bs3 = bsum_s[3*H_ + hh];
      #pragma unroll
      for (int r = 0; r < 8; ++r) {
        const int b  = mtile * 16 + r + (lhalf << 3);
        const float xv = xcol[b];
        const float iv = acc[0][p][r] + xv * wi0 + bs0;
        const float fv = acc[1][p][r] + xv * wi1 + bs1;
        const float gv = acc[2][p][r] + xv * wi2 + bs2;
        const float ov = acc[3][p][r] + xv * wi3 + bs3;
        const float cn = sigmf(fv) * creg[p * 8 + r] + sigmf(iv) * tanhf(gv);
        creg[p * 8 + r] = cn;                       // stop_gradient: fwd no-op
        const float hv = sigmf(ov) * tanhf(cn);
        h_bf[b * H_ + hh] = (__bf16)hv;             // bf16 h for next GEMM
        buf[((size_t)(t + 1) * B_ + b) * H_ + hh] = hv;  // append history slot
      }
    }
    __threadfence();
    __syncthreads();

    // ------------------ Phase B: s1[b], s2[t+1][b] ------------------
    {
      const int b = tid >> 3, seg = tid & 7;
      const float* hrow = buf + ((size_t)(t + 1) * B_ + b) * H_ + seg * 16;
      float a1 = 0.f, a2 = 0.f;
      #pragma unroll
      for (int e = 0; e < 16; ++e) {
        const float th = tanhf(hrow[e]);
        a1 += wt1_s[seg * 16 + e] * th;
        a2 += wt2_s[seg * 16 + e] * th;
      }
      #pragma unroll
      for (int off = 4; off > 0; off >>= 1) {
        a1 += __shfl_down(a1, off, 8);
        a2 += __shfl_down(a2, off, 8);
      }
      if (seg == 0) { s1s[b] = a1; s2g[(t + 1) * B_ + b] = a2; }
    }
    __threadfence();
    __syncthreads();

    // ------------------ Phase C: attention sum + FC head ------------------
    {
      const int b = tid >> 3, seg = tid & 7;
      const float s1v = s1s[b];
      float facc[16];
      #pragma unroll
      for (int e = 0; e < 16; ++e) facc[e] = 0.f;

      for (int j = 1; j <= t; ++j) {
        if (j + 8 <= t)   // pull upcoming history row toward the WGP
          __builtin_prefetch(buf + ((size_t)(j + 8) * B_ + b) * H_ + seg * 16, 0, 1);
        const float w = sigmf(s1v + s2g[j * B_ + b]);
        const float4* rp =
            (const float4*)(buf + ((size_t)j * B_ + b) * H_ + seg * 16);
        #pragma unroll
        for (int v = 0; v < 4; ++v) {
          const float4 d = rp[v];
          facc[4*v+0] += w * d.x; facc[4*v+1] += w * d.y;
          facc[4*v+2] += w * d.z; facc[4*v+3] += w * d.w;
        }
      }
      // add current h (f32, from the slot just written)
      const float4* hp =
          (const float4*)(buf + ((size_t)(t + 1) * B_ + b) * H_ + seg * 16);
      float o16[16];
      #pragma unroll
      for (int v = 0; v < 4; ++v) {
        const float4 d = hp[v];
        o16[4*v+0] = facc[4*v+0] + d.x; o16[4*v+1] = facc[4*v+1] + d.y;
        o16[4*v+2] = facc[4*v+2] + d.z; o16[4*v+3] = facc[4*v+3] + d.w;
      }
      float pc[C_];
      #pragma unroll
      for (int c = 0; c < C_; ++c) {
        float s = 0.f;
        #pragma unroll
        for (int e = 0; e < 16; ++e) s += o16[e] * fcW_s[c * H_ + seg * 16 + e];
        pc[c] = s;
      }
      #pragma unroll
      for (int c = 0; c < C_; ++c) {
        #pragma unroll
        for (int off = 4; off > 0; off >>= 1) pc[c] += __shfl_down(pc[c], off, 8);
      }
      if (seg == 0) {
        #pragma unroll
        for (int c = 0; c < C_; ++c)
          out[((size_t)b * T_ + t) * C_ + c] = pc[c] + fcb_s[c];
      }
    }
    __syncthreads();
  }
}

// ---------------------------------------------------------------------------
extern "C" void kernel_launch(void* const* d_in, const int* in_sizes, int n_in,
                              void* d_out, int out_size, void* d_ws, size_t ws_size,
                              hipStream_t stream) {
  const float* x    = (const float*)d_in[0];
  const float* W_ih = (const float*)d_in[1];
  const float* W_hh = (const float*)d_in[2];
  const float* b_ih = (const float*)d_in[3];
  const float* b_hh = (const float*)d_in[4];
  const float* w_t  = (const float*)d_in[5];
  const float* fc_W = (const float*)d_in[6];
  const float* fc_b = (const float*)d_in[7];
  float* out = (float*)d_out;

  char* ws = (char*)d_ws;
  float* s2g = (float*)ws;                       // 257*128*4 = 131584 B
  float* buf = (float*)(ws + 131584);            // 257*128*128*4 = 16.8 MB

  lstm_attn_persistent<<<1, 1024, LDS_TOTAL, stream>>>(
      x, W_ih, W_hh, b_ih, b_hh, w_t, fc_W, fc_b, s2g, buf, out);
}
